// RefineNet_14121852470058
// MI455X (gfx1250) — compile-verified, matrix operand order checked by reading, not run on hardware
//
#include <hip/hip_runtime.h>

typedef __attribute__((ext_vector_type(16))) _Float16 v16h;
typedef __attribute__((ext_vector_type(8)))  float    v8f;

#define CDIV(a,b) (((a)+(b)-1)/(b))

// ---------------- constants (reference shapes) ----------------
constexpr int BN = 2, HH = 128, WW = 128;
constexpr int HWc = HH * WW;          // 16384 (power of two -> shifts)
constexpr int LOG_HW = 14, LOG_W = 7;
constexpr int Pn  = BN * HWc;         // 32768 pixels total
constexpr int DIMF = 20, DIMC = 40;
constexpr int DI = 80;                // mamba inner dim = 2*DIMC
constexpr int NST = 16;               // DSTATE
constexpr int RNK = 3;                // ceil(40/16)
constexpr int XDW = RNK + 2 * NST;    // 35
// NHWC padded channel strides (multiples of 32 = WMMA K-chunk)
constexpr int CP32 = 32;              // for 6/15/18/20-channel maps
constexpr int CP64 = 64;              // for 40-channel maps (also mamba seq stride)
constexpr int CP96 = 96;              // for 80-channel maps
constexpr int UST = 96;               // u buffer stride   (K=80 -> Kpad 96)
constexpr int YST = 96;               // ys buffer stride  (K=80 -> Kpad 96)

// ---------------- device helpers ----------------
static __device__ inline int clampi(int v, int lo, int hi) { return v < lo ? lo : (v > hi ? hi : v); }

static __device__ inline void bilin(float ys, float xs,
                                    int& i00, int& i01, int& i10, int& i11,
                                    float& w00, float& w01, float& w10, float& w11) {
  float y0f = floorf(ys), x0f = floorf(xs);
  float wy = ys - y0f, wx = xs - x0f;
  int y0 = (int)y0f, x0 = (int)x0f, y1 = y0 + 1, x1 = x0 + 1;
  float v00 = (y0 >= 0 && y0 < HH && x0 >= 0 && x0 < WW) ? 1.f : 0.f;
  float v01 = (y0 >= 0 && y0 < HH && x1 >= 0 && x1 < WW) ? 1.f : 0.f;
  float v10 = (y1 >= 0 && y1 < HH && x0 >= 0 && x0 < WW) ? 1.f : 0.f;
  float v11 = (y1 >= 0 && y1 < HH && x1 >= 0 && x1 < WW) ? 1.f : 0.f;
  int cy0 = clampi(y0, 0, HH - 1), cy1 = clampi(y1, 0, HH - 1);
  int cx0 = clampi(x0, 0, WW - 1), cx1 = clampi(x1, 0, WW - 1);
  w00 = (1.f - wy) * (1.f - wx) * v00;
  w01 = (1.f - wy) * wx * v01;
  w10 = wy * (1.f - wx) * v10;
  w11 = wy * wx * v11;
  i00 = (cy0 << LOG_W) + cx0; i01 = (cy0 << LOG_W) + cx1;
  i10 = (cy1 << LOG_W) + cx0; i11 = (cy1 << LOG_W) + cx1;
}

static __device__ inline v16h cvt16(float4 a0, float4 a1, float4 a2, float4 a3) {
  v16h af;
  af[0] = (_Float16)a0.x; af[1] = (_Float16)a0.y; af[2] = (_Float16)a0.z; af[3] = (_Float16)a0.w;
  af[4] = (_Float16)a1.x; af[5] = (_Float16)a1.y; af[6] = (_Float16)a1.z; af[7] = (_Float16)a1.w;
  af[8] = (_Float16)a2.x; af[9] = (_Float16)a2.y; af[10] = (_Float16)a2.z; af[11] = (_Float16)a2.w;
  af[12] = (_Float16)a3.x; af[13] = (_Float16)a3.y; af[14] = (_Float16)a3.z; af[15] = (_Float16)a3.w;
  return af;
}

static __device__ inline float epi_act(float val, int act, float pa) {
  if      (act == 1) val = (val >= 0.f) ? val : pa * val;
  else if (act == 2) val = (val >= 0.f) ? val : 0.1f * val;
  else if (act == 3) val = fmaxf(val, 0.f);
  else if (act == 4) val = 0.5f * val * (1.f + erff(val * 0.70710678118654752f));
  else if (act == 5) val = fminf(fmaxf(val, -100.f), 100.f);
  else if (act == 6) val = fminf(fmaxf(val, 1e-8f), 1.f);
  return val;
}

// ---------------- weight packing: fragment-ready f16 [ntile][kchunk][lane][16] ------
// dense: weights row-major (N, K); Npad is a multiple of 32 (two tiles per wave)
__global__ void pack_dense(const float* __restrict__ w, _Float16* __restrict__ bp,
                           int O, int KK, int KC, int Npad) {
  int idx = blockIdx.x * blockDim.x + threadIdx.x;
  int total = Npad * KC * 32;
  if (idx >= total) return;
  int i = idx & 15; int t = idx >> 4;
  int lane = t & 31; t >>= 5;
  int kc = t % KC; int nt = t / KC;
  int kb = (lane < 16) ? 0 : 8;
  int ko = kb + (i < 8 ? i : 8 + i);
  int k = kc * 32 + ko;
  int n = nt * 16 + (lane & 15);
  float v = (k < KK && n < O) ? w[(size_t)n * KK + k] : 0.f;
  bp[idx] = (_Float16)v;
}
// conv3x3: OIHW weights, K ordered (tap, cin) with cin padded to Cpad
__global__ void pack_conv(const float* __restrict__ w, _Float16* __restrict__ bp,
                          int O, int Cin, int Cpad, int KC, int Npad) {
  int idx = blockIdx.x * blockDim.x + threadIdx.x;
  int total = Npad * KC * 32;
  if (idx >= total) return;
  int i = idx & 15; int t = idx >> 4;
  int lane = t & 31; t >>= 5;
  int kc = t % KC; int nt = t / KC;
  int kb = (lane < 16) ? 0 : 8;
  int ko = kb + (i < 8 ? i : 8 + i);
  int k = kc * 32 + ko;
  int tap = k / Cpad; int cin = k - tap * Cpad;
  int n = nt * 16 + (lane & 15);
  float v = (cin < Cin && n < O) ? w[((size_t)n * Cin + cin) * 9 + tap] : 0.f;
  bp[idx] = (_Float16)v;
}

// ---------------- dense WMMA GEMM: A[M x lda] row-major, pads zeroed ----------------
// two N-tiles per wave: one A fragment feeds two WMMAs
__global__ __launch_bounds__(128) void gemm_dense(
    const float* __restrict__ A, const _Float16* __restrict__ Bp,
    const float* __restrict__ bias, const float* __restrict__ pact,
    const float* __restrict__ Res, float* __restrict__ Y,
    int M, int N, int KC, int lda, int ldc, int act) {
  int lane = threadIdx.x & 31, wave = threadIdx.x >> 5;
  int mtile = blockIdx.x * 4 + wave, ntp = blockIdx.y;
  if (mtile * 16 >= M) return;
  int kb = (lane < 16) ? 0 : 8;
  int m_a = mtile * 16 + (lane & 15);
  const float* abase = A + (size_t)m_a * lda + kb;
  const _Float16* bptr0 = Bp + ((size_t)(ntp * 2) * KC * 32 + lane) * 16;
  const _Float16* bptr1 = bptr0 + (size_t)KC * 512;
  v8f acc0 = {}, acc1 = {};
  for (int kc = 0; kc < KC; ++kc) {
    const float* ap = abase + kc * 32;
    v16h af = cvt16(*(const float4*)(ap),      *(const float4*)(ap + 4),
                    *(const float4*)(ap + 16), *(const float4*)(ap + 20));
    v16h bf0 = *(const v16h*)(bptr0 + (size_t)kc * 512);
    v16h bf1 = *(const v16h*)(bptr1 + (size_t)kc * 512);
    acc0 = __builtin_amdgcn_wmma_f32_16x16x32_f16(false, af, false, bf0, (short)0, acc0, false, false);
    acc1 = __builtin_amdgcn_wmma_f32_16x16x32_f16(false, af, false, bf1, (short)0, acc1, false, false);
  }
  int mbase = mtile * 16 + ((lane >= 16) ? 8 : 0);
#pragma unroll
  for (int t = 0; t < 2; ++t) {
    int ncol = (ntp * 2 + t) * 16 + (lane & 15);
    if (ncol >= N) continue;
    v8f accv = (t == 0) ? acc0 : acc1;
    float bv = bias ? bias[ncol] : 0.f;
    float pa = pact ? pact[ncol] : 0.f;
#pragma unroll
    for (int v = 0; v < 8; ++v) {
      int m = mbase + v;
      float val = accv[v] + bv;
      size_t oidx = (size_t)m * ldc + ncol;
      if (Res) val += Res[oidx];
      Y[oidx] = epi_act(val, act, pa);
    }
  }
}

// ---------------- 3x3 conv WMMA GEMM: NHWC input, one predicate per tap -------------
template <int ONCHW>
__global__ __launch_bounds__(128) void gemm_conv3(
    const float* __restrict__ A, const _Float16* __restrict__ Bp,
    const float* __restrict__ bias, const float* __restrict__ pact,
    const float* __restrict__ Res, float* __restrict__ Y,
    int N, int CPC, int ldc, int act) {
  int lane = threadIdx.x & 31, wave = threadIdx.x >> 5;
  int mtile = blockIdx.x * 4 + wave, ntp = blockIdx.y;
  int kb = (lane < 16) ? 0 : 8;
  int m_a = mtile * 16 + (lane & 15);
  int bA = m_a >> LOG_HW;
  int hw = m_a & (HWc - 1);
  int hA = hw >> LOG_W, wA = hw & (WW - 1);
  int Cpad = CPC * 32;
  int KC = 9 * CPC;
  const _Float16* bptr0 = Bp + ((size_t)(ntp * 2) * KC * 32 + lane) * 16;
  const _Float16* bptr1 = bptr0 + (size_t)KC * 512;
  v8f acc0 = {}, acc1 = {};
  int kc = 0;
#pragma unroll
  for (int tap = 0; tap < 9; ++tap) {
    int iy = hA + tap / 3 - 1;
    int ix = wA + tap % 3 - 1;
    bool ok = (iy >= 0 && iy < HH && ix >= 0 && ix < WW);
    const float* base = A + (size_t)((bA << LOG_HW) + (iy << LOG_W) + ix) * Cpad + kb;
    for (int cc = 0; cc < CPC; ++cc, ++kc) {
      v16h af = {};
      if (ok) {
        const float* ap = base + cc * 32;
        af = cvt16(*(const float4*)(ap),      *(const float4*)(ap + 4),
                   *(const float4*)(ap + 16), *(const float4*)(ap + 20));
      }
      v16h bf0 = *(const v16h*)(bptr0 + (size_t)kc * 512);
      v16h bf1 = *(const v16h*)(bptr1 + (size_t)kc * 512);
      acc0 = __builtin_amdgcn_wmma_f32_16x16x32_f16(false, af, false, bf0, (short)0, acc0, false, false);
      acc1 = __builtin_amdgcn_wmma_f32_16x16x32_f16(false, af, false, bf1, (short)0, acc1, false, false);
    }
  }
  int mbase = mtile * 16 + ((lane >= 16) ? 8 : 0);
#pragma unroll
  for (int t = 0; t < 2; ++t) {
    int ncol = (ntp * 2 + t) * 16 + (lane & 15);
    if (ncol >= N) continue;
    v8f accv = (t == 0) ? acc0 : acc1;
    float bv = bias ? bias[ncol] : 0.f;
    float pa = pact ? pact[ncol] : 0.f;
#pragma unroll
    for (int v = 0; v < 8; ++v) {
      int m = mbase + v;
      float val = accv[v] + bv;
      size_t oidx;
      if (ONCHW) oidx = ((size_t)((m >> LOG_HW) * N + ncol) << LOG_HW) + (m & (HWc - 1));
      else       oidx = (size_t)m * ldc + ncol;
      if (Res) val += Res[oidx];
      Y[oidx] = epi_act(val, act, pa);
    }
  }
}

// ---------------- elementwise / utility kernels ----------------
__global__ void fill_k(float* __restrict__ p, float v, int n) {
  int i = blockIdx.x * blockDim.x + threadIdx.x;
  if (i < n) p[i] = v;
}
__global__ void mul_k(float* __restrict__ a, const float* __restrict__ b, int n) {
  int i = blockIdx.x * blockDim.x + threadIdx.x;
  if (i < n) a[i] *= b[i];
}
__global__ void gate_k(const float* __restrict__ f, const float* __restrict__ m,
                       float* __restrict__ o, int n) {
  int i = blockIdx.x * blockDim.x + threadIdx.x;
  if (i < n) o[i] = f[i] * (1.f / (1.f + expf(-m[i])));
}
__global__ void sftapply_k(const float* __restrict__ x0, const float* __restrict__ s,
                           const float* __restrict__ t, float* __restrict__ o, int n) {
  int i = blockIdx.x * blockDim.x + threadIdx.x;
  if (i < n) o[i] = x0[i] * (s[i] + 1.f) + t[i];
}
// NCHW source -> NHWC(Cd) dest at channel offset; grid.y = source channel
__global__ void cat_k(const float* __restrict__ src, float* __restrict__ dst,
                      int Cs, int Cd, int coff, float scale) {
  int m = blockIdx.x * blockDim.x + threadIdx.x;
  int c = blockIdx.y;
  if (m >= Pn) return;
  int b = m >> LOG_HW; int hw = m & (HWc - 1);
  dst[(size_t)m * Cd + coff + c] = src[((size_t)(b * Cs + c) << LOG_HW) + hw] * scale;
}
// NHWC(Cs) -> NHWC(Cd) channel-range copy; grid.y = source channel
__global__ void catn_k(const float* __restrict__ src, float* __restrict__ dst,
                       int Cs, int Cd, int coff) {
  int m = blockIdx.x * blockDim.x + threadIdx.x;
  int c = blockIdx.y;
  if (m >= Pn) return;
  dst[(size_t)m * Cd + coff + c] = src[(size_t)m * Cs + c];
}

__global__ void warp_k(const float* __restrict__ feat, const float* __restrict__ flow,
                       float* __restrict__ out, int C, int Cp) {
  int idx = blockIdx.x * blockDim.x + threadIdx.x;
  if (idx >= Pn) return;
  int b = idx >> LOG_HW, hw = idx & (HWc - 1), h = hw >> LOG_W, wq = hw & (WW - 1);
  float xs = (float)wq + flow[((size_t)(b * 2 + 0) << LOG_HW) + hw];
  float ys = (float)h  + flow[((size_t)(b * 2 + 1) << LOG_HW) + hw];
  int i00, i01, i10, i11; float w00, w01, w10, w11;
  bilin(ys, xs, i00, i01, i10, i11, w00, w01, w10, w11);
  size_t pb = (size_t)(b << LOG_HW);
  const float* f00 = feat + (pb + i00) * Cp; const float* f01 = feat + (pb + i01) * Cp;
  const float* f10 = feat + (pb + i10) * Cp; const float* f11 = feat + (pb + i11) * Cp;
  float* op = out + (size_t)idx * Cp;
  for (int c = 0; c < C; ++c)
    op[c] = w00 * f00[c] + w01 * f01[c] + w10 * f10[c] + w11 * f11[c];
}

// deformable conv on NHWC(CP32) maps, DIMF channels
__global__ void deform_k(const float* __restrict__ x, const float* __restrict__ off,
                         const float* __restrict__ w, const float* __restrict__ bias,
                         const float* __restrict__ res, float* __restrict__ y, int do_relu) {
  int idx = blockIdx.x * blockDim.x + threadIdx.x;
  if (idx >= Pn) return;
  int b = idx >> LOG_HW, hw = idx & (HWc - 1), h = hw >> LOG_W, wq = hw & (WW - 1);
  size_t pb = (size_t)(b << LOG_HW);
  float acc[DIMF];
#pragma unroll
  for (int o = 0; o < DIMF; ++o) acc[o] = bias[o];
  const float* offp = off + (size_t)idx * CP32;
  for (int k = 0; k < 9; ++k) {
    float ys = (float)h  + (float)(k / 3 - 1) + offp[2 * k];
    float xs = (float)wq + (float)(k % 3 - 1) + offp[2 * k + 1];
    int i00, i01, i10, i11; float w00, w01, w10, w11;
    bilin(ys, xs, i00, i01, i10, i11, w00, w01, w10, w11);
    const float* f00 = x + (pb + i00) * CP32; const float* f01 = x + (pb + i01) * CP32;
    const float* f10 = x + (pb + i10) * CP32; const float* f11 = x + (pb + i11) * CP32;
    for (int c = 0; c < DIMF; ++c) {
      float s = w00 * f00[c] + w01 * f01[c] + w10 * f10[c] + w11 * f11[c];
#pragma unroll
      for (int o = 0; o < DIMF; ++o) acc[o] += s * w[(size_t)(o * DIMF + c) * 9 + k];
    }
  }
  float* yp = y + (size_t)idx * CP32;
  const float* rp = res ? res + (size_t)idx * CP32 : nullptr;
  for (int o = 0; o < DIMF; ++o) {
    float v = acc[o];
    if (rp) v += rp[o];
    if (do_relu) v = fmaxf(v, 0.f);
    yp[o] = v;
  }
}

// channel layernorm on NHWC(CP64), 40 real channels (contiguous reduction)
__global__ void ln_k(const float* __restrict__ x, const float* __restrict__ g,
                     const float* __restrict__ be, float* __restrict__ y) {
  int idx = blockIdx.x * blockDim.x + threadIdx.x;
  if (idx >= Pn) return;
  const float* xb = x + (size_t)idx * CP64;
  float mu = 0.f;
  for (int c = 0; c < DIMC; ++c) mu += xb[c];
  mu /= (float)DIMC;
  float var = 0.f;
  for (int c = 0; c < DIMC; ++c) { float d = xb[c] - mu; var += d * d; }
  var /= (float)DIMC;
  float inv = rsqrtf(var + 1e-5f);
  float* yb = y + (size_t)idx * CP64;
  for (int c = 0; c < DIMC; ++c) yb[c] = (xb[c] - mu) * inv * g[c] + be[c];
}

// depthwise 1x3 / 3x1 on NHWC(CP64), writes directly into sequence order (flip-aware)
__global__ void dw_k(const float* __restrict__ x, const float* __restrict__ w,
                     const float* __restrict__ bias, float* __restrict__ seq,
                     int vertical, int flip) {
  int idx = blockIdx.x * blockDim.x + threadIdx.x;
  if (idx >= Pn * CP64) return;
  int c = idx & (CP64 - 1); int m0 = idx >> 6;
  if (c >= DIMC) return;
  int b = m0 >> LOG_HW; int hw = m0 & (HWc - 1);
  int h = hw >> LOG_W; int wq = hw & (WW - 1);
  int L = vertical ? HH : WW;
  int p = vertical ? h : wq;
  float acc = bias[c];
#pragma unroll
  for (int j = 0; j < 3; ++j) {
    int tp = p + j - 1;
    if (tp >= 0 && tp < L) {
      int src = flip ? (L - 1 - tp) : tp;
      int iy = vertical ? src : h;
      int ix = vertical ? wq : src;
      acc += x[(size_t)((b << LOG_HW) + (iy << LOG_W) + ix) * CP64 + c] * w[c * 3 + j];
    }
  }
  size_t mseq = vertical ? ((size_t)(b * WW + wq) * HH + h) : (size_t)m0;
  seq[mseq * CP64 + c] = acc;
}

// sequence (M x CP64) -> += NHWC(CP64); layout 0 hf, 1 hb, 2 vf, 3 vb
__global__ void scat_k(const float* __restrict__ mo, float* __restrict__ acc, int layout) {
  int idx = blockIdx.x * blockDim.x + threadIdx.x;
  if (idx >= Pn * CP64) return;
  int c = idx & (CP64 - 1); int m0 = idx >> 6;
  if (c >= DIMC) return;
  int b = m0 >> LOG_HW; int hw = m0 & (HWc - 1);
  int h = hw >> LOG_W; int wq = hw & (WW - 1);
  size_t m;
  if      (layout == 0) m = (size_t)m0;
  else if (layout == 1) m = (size_t)((b << LOG_HW) + (h << LOG_W) + (WW - 1 - wq));
  else if (layout == 2) m = (size_t)(b * WW + wq) * HH + h;
  else                  m = (size_t)(b * WW + wq) * HH + (HH - 1 - h);
  acc[idx] += mo[m * CP64 + c];
}

// causal depthwise conv1d(k=4, left pad 3) over xu = xz[:, :DI], + silu
__global__ void mamba_convu(const float* __restrict__ xz, const float* __restrict__ cw,
                            const float* __restrict__ cb, float* __restrict__ u,
                            int NBs, int Ls) {
  int idx = blockIdx.x * blockDim.x + threadIdx.x;
  int tot = NBs * Ls * DI;
  if (idx >= tot) return;
  int c = idx % DI; int m = idx / DI; int l = m % Ls; int bi = m / Ls;
  float acc = cb[c];
#pragma unroll
  for (int t = 0; t < 4; ++t) {
    int ll = l - 3 + t;
    if (ll >= 0) acc += xz[((size_t)(bi * Ls + ll)) * (2 * DI) + c] * cw[c * 4 + t];
  }
  u[(size_t)m * UST + c] = acc / (1.f + expf(-acc));   // silu
}

// dt = softplus(xdbl[:, :RNK] @ dt_w.T + dt_b)
__global__ void dt_k(const float* __restrict__ xdbl, const float* __restrict__ dt_w,
                     const float* __restrict__ dt_b, float* __restrict__ dt, int M) {
  int idx = blockIdx.x * blockDim.x + threadIdx.x;
  int tot = M * DI;
  if (idx >= tot) return;
  int c = idx % DI; int m = idx / DI;
  const float* xr = xdbl + (size_t)m * XDW;
  float v = dt_b[c];
#pragma unroll
  for (int j = 0; j < RNK; ++j) v += xr[j] * dt_w[c * RNK + j];
  dt[idx] = (v > 20.f) ? v : log1pf(expf(v));
}

// selective scan: blockDim 256 = 16 channels x 16 states; shuffle-reduce over 16 state lanes
__global__ __launch_bounds__(256) void scan_k(
    const float* __restrict__ u, const float* __restrict__ dt,
    const float* __restrict__ xdbl, const float* __restrict__ A_log,
    float* __restrict__ ys, int NBs, int Ls) {
  int bi = blockIdx.x / (DI / 16);
  int cg = blockIdx.x % (DI / 16);
  int n  = threadIdx.x & 15;
  int cl = threadIdx.x >> 4;
  int c  = cg * 16 + cl;
  float A = -expf(A_log[c * NST + n]);
  float h = 0.f;
  for (int l = 0; l < Ls; ++l) {
    size_t m = (size_t)bi * Ls + l;
    float dtv = dt[m * DI + c];
    float uv  = u[m * UST + c];
    float bm  = xdbl[m * XDW + RNK + n];
    float cm  = xdbl[m * XDW + RNK + NST + n];
    h = expf(dtv * A) * h + dtv * uv * bm;
    float p = h * cm;
    p += __shfl_xor(p, 8, 16);
    p += __shfl_xor(p, 4, 16);
    p += __shfl_xor(p, 2, 16);
    p += __shfl_xor(p, 1, 16);
    if (n == 0) ys[m * YST + c] = p;
  }
}

// y = (ys + u*D) * silu(z), z = xz[:, DI + c]
__global__ void ypost_k(float* __restrict__ ys, const float* __restrict__ u,
                        const float* __restrict__ xz, const float* __restrict__ D, int M) {
  int idx = blockIdx.x * blockDim.x + threadIdx.x;
  int tot = M * DI;
  if (idx >= tot) return;
  int c = idx % DI; int m = idx / DI;
  float z  = xz[(size_t)m * (2 * DI) + DI + c];
  float sz = z / (1.f + expf(-z));
  ys[(size_t)m * YST + c] = (ys[(size_t)m * YST + c] + u[(size_t)m * UST + c] * D[c]) * sz;
}

// ---------------- host-side parameter parsing (JAX pytree order: sorted dict keys) ----
struct PConv  { const float* b; const float* w; };                 // 'b','w'
struct PCrelu { const float* a; PConv c; };                        // 'a','c'
struct PSft   { PConv sc0, sc1, sh0, sh1; };
struct PDcrb  { PConv d1, d2, off; PSft sft1, sft2; };             // 'd1','d2','off','sft1','sft2'
struct PMamba { const float *A_log, *D, *cb, *cw, *dt_b, *dt_w, *in_w, *out_w, *xp_w; };
struct PVme   { PConv bch, bcv; const float* be; PConv fch, fcv; const float* g;
                PConv op; PMamba sbh, sbv, sfh, sfv; PConv to_x, to_z; };
struct PCon   { const float* a; PConv c; const float* ra; PCrelu rc1; PConv rc2; };

struct Cursor {
  void* const* d_in; int idx;
  const float* next() { return (const float*)d_in[idx++]; }
};
static PConv  rdConv (Cursor& p) { PConv c;  c.b = p.next(); c.w = p.next(); return c; }
static PCrelu rdCrelu(Cursor& p) { PCrelu r; r.a = p.next(); r.c = rdConv(p); return r; }
static PSft   rdSft  (Cursor& p) { PSft s; s.sc0 = rdConv(p); s.sc1 = rdConv(p);
                                   s.sh0 = rdConv(p); s.sh1 = rdConv(p); return s; }
static PDcrb  rdDcrb (Cursor& p) { PDcrb d; d.d1 = rdConv(p); d.d2 = rdConv(p); d.off = rdConv(p);
                                   d.sft1 = rdSft(p); d.sft2 = rdSft(p); return d; }
static PMamba rdMamba(Cursor& p) { PMamba m; m.A_log = p.next(); m.D = p.next(); m.cb = p.next();
                                   m.cw = p.next(); m.dt_b = p.next(); m.dt_w = p.next();
                                   m.in_w = p.next(); m.out_w = p.next(); m.xp_w = p.next(); return m; }
static PVme   rdVme  (Cursor& p) { PVme v; v.bch = rdConv(p); v.bcv = rdConv(p); v.be = p.next();
                                   v.fch = rdConv(p); v.fcv = rdConv(p); v.g = p.next();
                                   v.op = rdConv(p); v.sbh = rdMamba(p); v.sbv = rdMamba(p);
                                   v.sfh = rdMamba(p); v.sfv = rdMamba(p);
                                   v.to_x = rdConv(p); v.to_z = rdConv(p); return v; }
static PCon   rdCon  (Cursor& p) { PCon c; c.a = p.next(); c.c = rdConv(p); c.ra = p.next();
                                   c.rc1 = rdCrelu(p); c.rc2 = rdConv(p); return c; }

// ---------------- launch ----------------
extern "C" void kernel_launch(void* const* d_in, const int* in_sizes, int n_in,
                              void* d_out, int out_size, void* d_ws, size_t ws_size,
                              hipStream_t stream) {
  (void)in_sizes; (void)n_in; (void)out_size; (void)ws_size;

  const float* img0  = (const float*)d_in[0];
  const float* img1  = (const float*)d_in[1];
  const float* img2  = (const float*)d_in[2];
  const float* flow0 = (const float*)d_in[3];
  const float* flow2 = (const float*)d_in[4];
  const float* mask0 = (const float*)d_in[5];
  const float* mask2 = (const float*)d_in[6];
  const float* hdr   = (const float*)d_in[7];

  // params: top-level sorted keys: con, conv0, conv1, conv2, end, high, low, mam
  Cursor cur{d_in, 8};
  PCon   conP[2];  for (int i = 0; i < 2; ++i) conP[i]  = rdCon(cur);
  PCrelu c0P[2];   for (int i = 0; i < 2; ++i) c0P[i]   = rdCrelu(cur);
  PCrelu c1P[2];   for (int i = 0; i < 2; ++i) c1P[i]   = rdCrelu(cur);
  PCrelu c2P[2];   for (int i = 0; i < 2; ++i) c2P[i]   = rdCrelu(cur);
  PConv  endP      = rdConv(cur);
  PDcrb  highP[2]; for (int i = 0; i < 2; ++i) highP[i] = rdDcrb(cur);
  PDcrb  lowP[2];  for (int i = 0; i < 2; ++i) lowP[i]  = rdDcrb(cur);
  PVme   mamP[2][2];
  for (int i = 0; i < 2; ++i) for (int j = 0; j < 2; ++j) mamP[i][j] = rdVme(cur);

  // workspace arena (all chunks multiples of Pn floats -> 16B/32B aligned)
  float* ws = (float*)d_ws;
  size_t wo = 0;
  _Float16* bstage = (_Float16*)(ws + wo); wo += 65536;   // 128K halfs for packed B
  auto FA = [&](size_t n) { float* p = ws + wo; wo += n; return p; };

  float* t32a   = FA((size_t)CP32 * Pn);  float* t32b   = FA((size_t)CP32 * Pn);
  float* x1cat  = FA((size_t)CP32 * Pn);
  float* feat0A = FA((size_t)CP32 * Pn);  float* feat0B = FA((size_t)CP32 * Pn);
  float* feat2A = FA((size_t)CP32 * Pn);  float* feat2B = FA((size_t)CP32 * Pn);
  float* offb   = FA((size_t)CP32 * Pn);
  float* sbuf   = FA((size_t)CP32 * Pn);  float* tbuf   = FA((size_t)CP32 * Pn);
  float* feabuf = FA((size_t)CP32 * Pn);  float* fea2   = FA((size_t)CP32 * Pn);
  float* cat80  = FA((size_t)CP96 * Pn);
  float* feat1  = FA((size_t)CP64 * Pn);  float* featC  = FA((size_t)CP64 * Pn);
  float* t64a   = FA((size_t)CP64 * Pn);  float* featD  = FA((size_t)CP64 * Pn);
  float* v1     = FA((size_t)CP64 * Pn);  float* v2     = FA((size_t)CP64 * Pn);
  float* xnb    = FA((size_t)CP64 * Pn);  float* xpb    = FA((size_t)CP64 * Pn);
  float* zgb    = FA((size_t)CP64 * Pn);  float* accb   = FA((size_t)CP64 * Pn);
  float* mob    = FA((size_t)CP64 * Pn);
  float* seqb   = FA((size_t)CP64 * Pn);
  float* xzb    = FA((size_t)160 * Pn);
  float* ubuf   = FA((size_t)UST * Pn);
  float* xdblb  = FA((size_t)35 * Pn);
  float* dtbuf  = FA((size_t)80 * Pn);
  float* ysb    = FA((size_t)YST * Pn);

  // zero the whole arena once per launch: all pad channels stay zero for the call
  fill_k<<<CDIV((int)wo, 256), 256, 0, stream>>>(ws, 0.f, (int)wo);

  // dense GEMM (also used for every 1x1 conv): A [M x lda] NHWC-style, pads zeroed
  auto run_dense = [&](const float* A, const float* Wt, int M, int N, int K, int lda, int ldc,
                       float* Y, int act, const float* bias, const float* pa, const float* res) {
    int Kpad = CDIV(K, 32) * 32, Npad = CDIV(N, 32) * 32, KC = Kpad / 32;
    pack_dense<<<CDIV(Kpad * Npad, 256), 256, 0, stream>>>(Wt, bstage, N, K, KC, Npad);
    dim3 g(CDIV(CDIV(M, 16), 4), Npad / 32);
    gemm_dense<<<g, 128, 0, stream>>>(A, bstage, bias, pa, res, Y, M, N, KC, lda, ldc, act);
  };

  auto run_conv3 = [&](const float* X, const PConv& pc, int Cin, int Cpad_in, int N, int ldc,
                       float* Y, int act, const float* pa, const float* res, bool onchw) {
    int CPC = Cpad_in / 32, KC = 9 * CPC;
    int Npad = CDIV(N, 32) * 32;
    pack_conv<<<CDIV(KC * 32 * Npad, 256), 256, 0, stream>>>(pc.w, bstage, N, Cin, Cpad_in, KC, Npad);
    dim3 g(Pn / 64, Npad / 32);
    if (onchw)
      gemm_conv3<1><<<g, 128, 0, stream>>>(X, bstage, pc.b, pa, res, Y, N, CPC, ldc, act);
    else
      gemm_conv3<0><<<g, 128, 0, stream>>>(X, bstage, pc.b, pa, res, Y, N, CPC, ldc, act);
  };

  auto run_mamba = [&](const float* seq, const PMamba& pm, int NBs, int Ls, float* out) {
    int M = NBs * Ls;
    run_dense(seq, pm.in_w, M, 2 * DI, DIMC, CP64, 2 * DI, xzb, 0, nullptr, nullptr, nullptr);
    mamba_convu<<<CDIV(M * DI, 256), 256, 0, stream>>>(xzb, pm.cw, pm.cb, ubuf, NBs, Ls);
    run_dense(ubuf, pm.xp_w, M, XDW, DI, UST, XDW, xdblb, 0, nullptr, nullptr, nullptr);
    dt_k<<<CDIV(M * DI, 256), 256, 0, stream>>>(xdblb, pm.dt_w, pm.dt_b, dtbuf, M);
    scan_k<<<dim3(NBs * (DI / 16)), 256, 0, stream>>>(ubuf, dtbuf, xdblb, pm.A_log, ysb, NBs, Ls);
    ypost_k<<<CDIV(M * DI, 256), 256, 0, stream>>>(ysb, ubuf, xzb, pm.D, M);
    run_dense(ysb, pm.out_w, M, DIMC, DI, YST, CP64, out, 0, nullptr, nullptr, nullptr);
  };

  auto run_vme = [&](const float* X, const PVme& pv, float* OUT) {
    ln_k<<<CDIV(Pn, 256), 256, 0, stream>>>(X, pv.g, pv.be, xnb);
    run_dense(xnb, pv.to_x.w, Pn, DIMC, DIMC, CP64, CP64, xpb, 0, pv.to_x.b, nullptr, nullptr);
    run_dense(xnb, pv.to_z.w, Pn, DIMC, DIMC, CP64, CP64, zgb, 4, pv.to_z.b, nullptr, nullptr);
    fill_k<<<CDIV(CP64 * Pn, 256), 256, 0, stream>>>(accb, 0.f, CP64 * Pn);
    const PConv*  dwp[4] = {&pv.fch, &pv.bch, &pv.fcv, &pv.bcv};
    const PMamba* mp[4]  = {&pv.sfh, &pv.sbh, &pv.sfv, &pv.sbv};
    for (int d = 0; d < 4; ++d) {
      int vertical = (d >= 2), flip = (d & 1);
      dw_k<<<CDIV(CP64 * Pn, 256), 256, 0, stream>>>(xpb, dwp[d]->w, dwp[d]->b, seqb, vertical, flip);
      int NBs = vertical ? BN * WW : BN;
      int Ls  = vertical ? HH : HWc;
      run_mamba(seqb, *mp[d], NBs, Ls, mob);
      scat_k<<<CDIV(CP64 * Pn, 256), 256, 0, stream>>>(mob, accb, d);
    }
    mul_k<<<CDIV(CP64 * Pn, 256), 256, 0, stream>>>(accb, zgb, CP64 * Pn);
    run_dense(accb, pv.op.w, Pn, DIMC, DIMC, CP64, CP64, OUT, 5, pv.op.b, nullptr, X);
  };

  auto run_sft = [&](const float* x0, const float* x1, const PSft& ps, float* out) {
    run_dense(x1, ps.sc0.w, Pn, DIMF, DIMC, CP64, CP32, t32a, 2, ps.sc0.b, nullptr, nullptr);
    run_dense(t32a, ps.sc1.w, Pn, DIMF, DIMF, CP32, CP32, sbuf, 0, ps.sc1.b, nullptr, nullptr);
    run_dense(x1, ps.sh0.w, Pn, DIMF, DIMC, CP64, CP32, t32a, 2, ps.sh0.b, nullptr, nullptr);
    run_dense(t32a, ps.sh1.w, Pn, DIMF, DIMF, CP32, CP32, tbuf, 0, ps.sh1.b, nullptr, nullptr);
    sftapply_k<<<CDIV(CP32 * Pn, 256), 256, 0, stream>>>(x0, sbuf, tbuf, out, CP32 * Pn);
  };

  auto run_dcrb = [&](const float* x0, const float* x1, const PDcrb& pd, float* xout) {
    run_conv3(x0, pd.off, DIMF, CP32, 18, CP32, offb, 0, nullptr, nullptr, false);
    run_sft(x0, x1, pd.sft1, feabuf);
    deform_k<<<CDIV(Pn, 256), 256, 0, stream>>>(feabuf, offb, pd.d1.w, pd.d1.b, nullptr, fea2, 1);
    run_sft(fea2, x1, pd.sft2, feabuf);
    deform_k<<<CDIV(Pn, 256), 256, 0, stream>>>(feabuf, offb, pd.d2.w, pd.d2.b, x0, xout, 0);
  };

  // ---------------- forward pass ----------------
  // feat0 = warp(crelu(crelu(img0)))
  cat_k<<<dim3(CDIV(Pn, 256), 6), 256, 0, stream>>>(img0, t32a, 6, CP32, 0, 1.f);
  run_conv3(t32a, c0P[0].c, 6, CP32, DIMF, CP32, t32b, 1, c0P[0].a, nullptr, false);
  run_conv3(t32b, c0P[1].c, DIMF, CP32, DIMF, CP32, t32a, 1, c0P[1].a, nullptr, false);
  warp_k<<<CDIV(Pn, 256), 256, 0, stream>>>(t32a, flow0, feat0A, DIMF, CP32);

  // x1 = cat([img1, flow0/20, flow2/20, mask0, mask2, hdr]) -> NHWC32
  cat_k<<<dim3(CDIV(Pn, 256), 6), 256, 0, stream>>>(img1,  x1cat, 6, CP32, 0,  1.f);
  cat_k<<<dim3(CDIV(Pn, 256), 2), 256, 0, stream>>>(flow0, x1cat, 2, CP32, 6,  0.05f);
  cat_k<<<dim3(CDIV(Pn, 256), 2), 256, 0, stream>>>(flow2, x1cat, 2, CP32, 8,  0.05f);
  cat_k<<<dim3(CDIV(Pn, 256), 1), 256, 0, stream>>>(mask0, x1cat, 1, CP32, 10, 1.f);
  cat_k<<<dim3(CDIV(Pn, 256), 1), 256, 0, stream>>>(mask2, x1cat, 1, CP32, 11, 1.f);
  cat_k<<<dim3(CDIV(Pn, 256), 3), 256, 0, stream>>>(hdr,   x1cat, 3, CP32, 12, 1.f);
  run_conv3(x1cat, c1P[0].c, 15, CP32, DIMC, CP64, t64a, 1, c1P[0].a, nullptr, false);
  run_conv3(t64a, c1P[1].c, DIMC, CP64, DIMC, CP64, feat1, 1, c1P[1].a, nullptr, false);

  // feat2 = warp(crelu(crelu(img2)))
  cat_k<<<dim3(CDIV(Pn, 256), 6), 256, 0, stream>>>(img2, t32a, 6, CP32, 0, 1.f);
  run_conv3(t32a, c2P[0].c, 6, CP32, DIMF, CP32, t32b, 1, c2P[0].a, nullptr, false);
  run_conv3(t32b, c2P[1].c, DIMF, CP32, DIMF, CP32, t32a, 1, c2P[1].a, nullptr, false);
  warp_k<<<CDIV(Pn, 256), 256, 0, stream>>>(t32a, flow2, feat2A, DIMF, CP32);

  float* f0 = feat0A; float* f0alt = feat0B;
  float* f2 = feat2A; float* f2alt = feat2B;
  for (int i = 0; i < 2; ++i) {
    run_dcrb(f0, feat1, lowP[i],  f0alt); { float* t = f0; f0 = f0alt; f0alt = t; }
    run_dcrb(f2, feat1, highP[i], f2alt); { float* t = f2; f2 = f2alt; f2alt = t; }
    catn_k<<<dim3(CDIV(Pn, 256), DIMF), 256, 0, stream>>>(f0,    cat80, CP32, CP96, 0);
    catn_k<<<dim3(CDIV(Pn, 256), DIMC), 256, 0, stream>>>(feat1, cat80, CP64, CP96, 20);
    catn_k<<<dim3(CDIV(Pn, 256), DIMF), 256, 0, stream>>>(f2,    cat80, CP32, CP96, 60);
    run_conv3(cat80, conP[i].c, 80, CP96, DIMC, CP64, featC, 1, conP[i].a, nullptr, false);
    run_conv3(featC, conP[i].rc1.c, DIMC, CP64, DIMC, CP64, t64a, 1, conP[i].rc1.a, nullptr, false);
    run_conv3(t64a, conP[i].rc2, DIMC, CP64, DIMC, CP64, featD, 1, conP[i].ra, featC, false);
    run_vme(featD, mamP[i][0], v1);
    run_vme(v1, mamP[i][1], v2);
    gate_k<<<CDIV(CP64 * Pn, 256), 256, 0, stream>>>(featD, v2, feat1, CP64 * Pn);
  }

  // out = clip(hdr + conv3x3(feat1 -> 3ch), 1e-8, 1)  (NCHW output to match harness)
  run_conv3(feat1, endP, DIMC, CP64, 3, 0, (float*)d_out, 6, nullptr, hdr, true);
}
